// CausalSelfAttention_66984309948568
// MI455X (gfx1250) — matrix operand order repaired and checked by reading, not compiled, not run on hardware
//
#include <hip/hip_runtime.h>
#include <hip/hip_bf16.h>

typedef _Float16 h16;
typedef __attribute__((ext_vector_type(16))) _Float16 v16h;
typedef __attribute__((ext_vector_type(8)))  _Float16 v8h;
typedef __attribute__((ext_vector_type(8)))  float    v8f;
typedef int v4i_ __attribute__((vector_size(16)));

#define T_LEN   4096
#define DIM_    1024
#define NHEAD   8
#define HDIM_   1024
#define HD_     128
#define SCALE_  0.12f
#define EPS_    1.1920928955078125e-07f

#define WMMA16(a,b,c) __builtin_amdgcn_wmma_f32_16x16x32_f16(false,(a),false,(b),(short)0,(c),false,false)

// ---- CDNA5 async global->LDS copy (ASYNCcnt path), with sync fallback ----
#if __has_builtin(__builtin_amdgcn_global_load_async_to_lds_b128) && \
    __has_builtin(__builtin_amdgcn_s_wait_asynccnt)
#define HAS_ASYNC_LDS 1
#else
#define HAS_ASYNC_LDS 0
#warning "gfx1250 async-to-LDS builtins not found: falling back to sync LDS staging"
#endif

__device__ inline void cpa16(h16* l, const h16* g) {
#if HAS_ASYNC_LDS
  // signature (from hipcc diagnostic): (v4i __device__*, v4i __shared__*, Imm offset, Imm cpol)
  __builtin_amdgcn_global_load_async_to_lds_b128(
      (__attribute__((address_space(1))) v4i_*)(v4i_*)g,
      (__attribute__((address_space(3))) v4i_*)(v4i_*)l, 0, 0);
#else
  *(v8h*)l = *(const v8h*)g;
#endif
}
#if HAS_ASYNC_LDS
#define AWAIT(n) __builtin_amdgcn_s_wait_asynccnt(n)
#else
#define AWAIT(n)
#endif

__device__ inline v16h cat8(v8h lo, v8h hi) {
  return __builtin_shufflevector(lo, hi, 0,1,2,3,4,5,6,7,8,9,10,11,12,13,14,15);
}
__device__ inline v8f zero8() {
  v8f z = {0.f,0.f,0.f,0.f,0.f,0.f,0.f,0.f};
  return z;
}
__device__ inline float red_sum16(float v) {
  v += __shfl_xor(v, 1, 32); v += __shfl_xor(v, 2, 32);
  v += __shfl_xor(v, 4, 32); v += __shfl_xor(v, 8, 32);
  return v;
}
__device__ inline float red_max16(float v) {
  v = fmaxf(v, __shfl_xor(v, 1, 32)); v = fmaxf(v, __shfl_xor(v, 2, 32));
  v = fmaxf(v, __shfl_xor(v, 4, 32)); v = fmaxf(v, __shfl_xor(v, 8, 32));
  return v;
}

// ---------------- fp32 -> fp16 convert ----------------
__global__ void cvt_f32_f16(const float* __restrict__ s, h16* __restrict__ d, int n) {
  int i = blockIdx.x * blockDim.x + threadIdx.x;
  if (i < n) d[i] = (h16)s[i];
}

// ---------------- WMMA GEMM core: 64(M) x 128(N), A*B^T, double-buffered LDS ----------------
// A: [Mtot][K] f16 row-major; B: [Ntot][K] f16 row-major. 128 threads = 4 waves.
#define AST_ (64 * 72)
#define BST_ (128 * 72)

__device__ inline void gemm_issue(const h16* __restrict__ A, const h16* __restrict__ B,
                                  int K, int m0, int n0, int k0,
                                  h16* ab, h16* bb) {
  const int tid = threadIdx.x;
#pragma unroll
  for (int j = 0; j < 4; ++j) {                 // A tile: 64 x 64 halves
    int c = tid + 128 * j; int r = c >> 3, cc = (c & 7) << 3;
    cpa16(ab + r * 72 + cc, A + (size_t)(m0 + r) * K + k0 + cc);
  }
#pragma unroll
  for (int j = 0; j < 8; ++j) {                 // B tile: 128 x 64 halves
    int c = tid + 128 * j; int r = c >> 3, cc = (c & 7) << 3;
    cpa16(bb + r * 72 + cc, B + (size_t)(n0 + r) * K + k0 + cc);
  }
}

__device__ inline void gemm_core(const h16* __restrict__ A, const h16* __restrict__ B,
                                 int K, int m0, int n0,
                                 h16* ab, h16* bb, v8f acc[8]) {
  const int tid = threadIdx.x;
  const int lane = tid & 31, wid = tid >> 5;
  const int lm = lane & 15, lh = lane >> 4;
#pragma unroll
  for (int i = 0; i < 8; ++i) acc[i] = zero8();

  const int nst = K >> 6;                        // 64-wide K stages
  gemm_issue(A, B, K, m0, n0, 0, ab, bb);        // preload stage 0
  for (int i = 0; i < nst; ++i) {
    const int cur = i & 1;
    if (i + 1 < nst) {                           // copy for stage i+1 overlaps compute of i
      gemm_issue(A, B, K, m0, n0, (i + 1) << 6, ab + (cur ^ 1) * AST_, bb + (cur ^ 1) * BST_);
      AWAIT(12);                                 // 12 chunks in flight for i+1; stage i done
    } else {
      AWAIT(0);
    }
    __syncthreads();
    const h16* abc = ab + cur * AST_;
    const h16* bbc = bb + cur * BST_;
#pragma unroll
    for (int kk = 0; kk < 2; ++kk) {
      const h16* ar = abc + (wid * 16 + lm) * 72 + kk * 32 + 8 * lh;
      v16h af = cat8(*(const v8h*)ar, *(const v8h*)(ar + 16));
#pragma unroll
      for (int nt = 0; nt < 8; ++nt) {
        const h16* br = bbc + (nt * 16 + lm) * 72 + kk * 32 + 8 * lh;
        v16h bf = cat8(*(const v8h*)br, *(const v8h*)(br + 16));
        acc[nt] = WMMA16(af, bf, acc[nt]);
      }
    }
    __syncthreads();                             // all waves done reading buf[cur]
  }
}

// ---------------- QKV GEMM + fused RMSNorm/RoPE (q,k) or lambda-combine (v) ----------------
__global__ __launch_bounds__(128) void qkv_kernel(
    const h16* __restrict__ xh, const h16* __restrict__ wh,
    const float* __restrict__ ve, const float* __restrict__ lambdas,
    h16* __restrict__ qh, h16* __restrict__ kh, h16* __restrict__ vt) {
  __shared__ __align__(16) h16 ab[2 * AST_];
  __shared__ __align__(16) h16 bb[2 * BST_];
  const int m0 = blockIdx.x * 64;
  const int ntile = blockIdx.y;
  v8f acc[8];
  gemm_core(xh, wh, DIM_, m0, ntile * 128, ab, bb, acc);

  const int lane = threadIdx.x & 31, wid = threadIdx.x >> 5;
  const int lm = lane & 15, lh = lane >> 4;
  const int which = ntile >> 3, head = ntile & 7;
  const int tbase = m0 + wid * 16 + 8 * lh;      // row = tbase + r

  if (which < 2) {
    // RMS norm over the 128-wide head (per row)
#pragma unroll
    for (int r = 0; r < 8; ++r) {
      float ss = 0.f;
#pragma unroll
      for (int nt = 0; nt < 8; ++nt) ss += acc[nt][r] * acc[nt][r];
      ss = red_sum16(ss);
      float sc = rsqrtf(ss * (1.0f / 128.0f) + EPS_);
#pragma unroll
      for (int nt = 0; nt < 8; ++nt) acc[nt][r] *= sc;
    }
    // RoPE: dim n pairs with n+64 (tile j <-> j+4, same lane). Angles zero for n>=32.
#pragma unroll
    for (int j = 0; j < 4; ++j) {
      int n1 = 16 * j + lm;
      float ang = (n1 < 32) ? exp2f(-10.0f * (float)n1 * (1.0f / 31.0f)) : 0.0f;
#pragma unroll
      for (int r = 0; r < 8; ++r) {
        float th = (float)(tbase + r) * ang;
        float c = __cosf(th), s = __sinf(th);
        float x1 = acc[j][r], x2 = acc[j + 4][r];
        acc[j][r]     = x1 * c + x2 * s;
        acc[j + 4][r] = x2 * c - x1 * s;
      }
    }
    h16* dst = (which == 0) ? qh : kh;
#pragma unroll
    for (int nt = 0; nt < 8; ++nt)
#pragma unroll
      for (int r = 0; r < 8; ++r)
        dst[((size_t)(tbase + r) * NHEAD + head) * HD_ + 16 * nt + lm] = (h16)acc[nt][r];
  } else {
    const float l0 = lambdas[0], l1 = lambdas[1];
#pragma unroll
    for (int nt = 0; nt < 8; ++nt) {
      int n = 16 * nt + lm;
#pragma unroll
      for (int r = 0; r < 8; ++r) {
        int t = tbase + r;
        float vv = l0 * acc[nt][r] + l1 * ve[(size_t)t * HDIM_ + head * HD_ + n];
        vt[((size_t)head * HD_ + n) * T_LEN + t] = (h16)vv;  // V transposed: [h][d][T]
      }
    }
  }
}

// ---------------- Flash attention: grid (64, 8) = (64-row q block, head) ----------------
#define KST_ (64 * 136)
#define VST_ (128 * 72)

__device__ inline void attn_issue(const h16* __restrict__ kh, const h16* __restrict__ vt,
                                  int h, int s0, h16* kb, h16* vb) {
  const int tid = threadIdx.x;
#pragma unroll
  for (int j = 0; j < 8; ++j) {                  // K tile: 64 keys x 128 dims
    int c = tid + 128 * j; int r = c >> 4, cc = (c & 15) << 3;
    cpa16(kb + r * 136 + cc, kh + ((size_t)(s0 + r) * NHEAD + h) * HD_ + cc);
  }
#pragma unroll
  for (int j = 0; j < 8; ++j) {                  // V^T tile: 128 dims x 64 keys
    int c = tid + 128 * j; int d = c >> 3, cc = (c & 7) << 3;
    cpa16(vb + d * 72 + cc, vt + ((size_t)h * HD_ + d) * T_LEN + s0 + cc);
  }
}

__global__ __launch_bounds__(128) void attn_kernel(
    const h16* __restrict__ qh, const h16* __restrict__ kh,
    const h16* __restrict__ vt, h16* __restrict__ y) {
  __shared__ __align__(16) h16 kb[2 * KST_];
  __shared__ __align__(16) h16 vb[2 * VST_];
  __shared__ __align__(16) h16 pbuf[4][16][72];

  const int tid = threadIdx.x;
  const int lane = tid & 31, wid = tid >> 5;
  const int lm = lane & 15, lh = lane >> 4;
  const int qb = blockIdx.x, h = blockIdx.y;
  const int qt0 = qb * 64 + wid * 16;            // this wave's 16 q rows

  // Q fragments held in registers for the whole pass: 16 rows x 128 dims
  v16h qf[4];
  {
    const h16* qrow = &qh[((size_t)(qt0 + lm) * NHEAD + h) * HD_];
#pragma unroll
    for (int kk = 0; kk < 4; ++kk) {
      const h16* p = qrow + kk * 32 + 8 * lh;
      qf[kk] = cat8(*(const v8h*)p, *(const v8h*)(p + 16));
    }
  }

  v8f o[8];
#pragma unroll
  for (int i = 0; i < 8; ++i) o[i] = zero8();
  float mrow[8], lrow[8];
#pragma unroll
  for (int r = 0; r < 8; ++r) { mrow[r] = -3.0e38f; lrow[r] = 0.f; }

  const int ns = qb + 1;                          // number of 64-key stages (uniform in block)
  attn_issue(kh, vt, h, 0, kb, vb);               // preload stage 0
  for (int i = 0; i < ns; ++i) {
    const int cur = i & 1;
    const int s0 = i * 64;
    if (i + 1 < ns) {                             // stage i+1 copy overlaps compute of i
      attn_issue(kh, vt, h, s0 + 64, kb + (cur ^ 1) * KST_, vb + (cur ^ 1) * VST_);
      AWAIT(16);
    } else {
      AWAIT(0);
    }
    __syncthreads();
    const h16* kbc = kb + cur * KST_;
    const h16* vbc = vb + cur * VST_;

    // S = Q K^T : 16 x 64, accumulated over head_dim 128
    v8f sacc[4];
#pragma unroll
    for (int st = 0; st < 4; ++st) sacc[st] = zero8();
#pragma unroll
    for (int kk = 0; kk < 4; ++kk) {
#pragma unroll
      for (int st = 0; st < 4; ++st) {
        const h16* br = kbc + (16 * st + lm) * 136 + kk * 32 + 8 * lh;
        v16h bf = cat8(*(const v8h*)br, *(const v8h*)(br + 16));
        sacc[st] = WMMA16(qf[kk], bf, sacc[st]);
      }
    }

    // scale + causal mask
    const int trow = qt0 + 8 * lh;
#pragma unroll
    for (int st = 0; st < 4; ++st) {
      int s = s0 + 16 * st + lm;
#pragma unroll
      for (int r = 0; r < 8; ++r) {
        float v = sacc[st][r] * SCALE_;
        if (s > trow + r) v = -1.0e30f;
        sacc[st][r] = v;
      }
    }

    // online softmax update
    float alpha[8];
#pragma unroll
    for (int r = 0; r < 8; ++r) {
      float mx = fmaxf(fmaxf(sacc[0][r], sacc[1][r]), fmaxf(sacc[2][r], sacc[3][r]));
      mx = red_max16(mx);
      float mnew = fmaxf(mrow[r], mx);
      alpha[r] = __expf(mrow[r] - mnew);
      mrow[r] = mnew;
    }
#pragma unroll
    for (int st = 0; st < 4; ++st)
#pragma unroll
      for (int r = 0; r < 8; ++r)
        sacc[st][r] = __expf(sacc[st][r] - mrow[r]);
#pragma unroll
    for (int r = 0; r < 8; ++r) {
      float rs = sacc[0][r] + sacc[1][r] + sacc[2][r] + sacc[3][r];
      rs = red_sum16(rs);
      lrow[r] = lrow[r] * alpha[r] + rs;
    }
#pragma unroll
    for (int nt = 0; nt < 8; ++nt)
#pragma unroll
      for (int r = 0; r < 8; ++r) o[nt][r] *= alpha[r];

    // P (C-layout f32) -> wave-private LDS -> A-fragment f16 (same-wave DS is in-order)
#pragma unroll
    for (int st = 0; st < 4; ++st)
#pragma unroll
      for (int r = 0; r < 8; ++r)
        pbuf[wid][r + 8 * lh][16 * st + lm] = (h16)sacc[st][r];

    // O += P V
#pragma unroll
    for (int kk2 = 0; kk2 < 2; ++kk2) {
      const h16* pr = &pbuf[wid][lm][kk2 * 32 + 8 * lh];
      v16h pf = cat8(*(const v8h*)pr, *(const v8h*)(pr + 16));
#pragma unroll
      for (int nt = 0; nt < 8; ++nt) {
        const h16* vr = vbc + (16 * nt + lm) * 72 + kk2 * 32 + 8 * lh;
        v16h vf = cat8(*(const v8h*)vr, *(const v8h*)(vr + 16));
        o[nt] = WMMA16(pf, vf, o[nt]);
      }
    }
    __syncthreads();                              // buf[cur] free for stage i+2
  }

  // epilogue: O / l -> y (f16)
#pragma unroll
  for (int nt = 0; nt < 8; ++nt)
#pragma unroll
    for (int r = 0; r < 8; ++r) {
      int t = qt0 + 8 * lh + r;
      y[(size_t)t * HDIM_ + h * HD_ + 16 * nt + lm] = (h16)(o[nt][r] / lrow[r]);
    }
}

// ---------------- output projection GEMM ----------------
__global__ __launch_bounds__(128) void proj_kernel(
    const h16* __restrict__ yh, const h16* __restrict__ pwh, float* __restrict__ out) {
  __shared__ __align__(16) h16 ab[2 * AST_];
  __shared__ __align__(16) h16 bb[2 * BST_];
  const int m0 = blockIdx.x * 64, n0 = blockIdx.y * 128;
  v8f acc[8];
  gemm_core(yh, pwh, HDIM_, m0, n0, ab, bb, acc);

  const int lane = threadIdx.x & 31, wid = threadIdx.x >> 5;
  const int lm = lane & 15, lh = lane >> 4;
  const int tbase = m0 + wid * 16 + 8 * lh;
#pragma unroll
  for (int nt = 0; nt < 8; ++nt)
#pragma unroll
    for (int r = 0; r < 8; ++r)
      out[(size_t)(tbase + r) * DIM_ + n0 + 16 * nt + lm] = acc[nt][r];
}

extern "C" void kernel_launch(void* const* d_in, const int* in_sizes, int n_in,
                              void* d_out, int out_size, void* d_ws, size_t ws_size,
                              hipStream_t stream) {
  (void)in_sizes; (void)n_in; (void)out_size; (void)ws_size;
  const float* x       = (const float*)d_in[0];   // [T][DIM]
  const float* ve      = (const float*)d_in[1];   // [T][HDIM]
  const float* qkv_w   = (const float*)d_in[2];   // [3*HDIM][DIM]
  const float* lambdas = (const float*)d_in[3];   // [2]
  const float* proj_w  = (const float*)d_in[4];   // [DIM][HDIM]

  char* ws = (char*)d_ws;
  h16* xh  = (h16*)(ws);                              // 4096*1024 f16 =  8 MiB
  h16* wh  = (h16*)(ws + 8388608);                    // 3072*1024 f16 =  6 MiB
  h16* pwh = (h16*)(ws + 8388608 + 6291456);          // 1024*1024 f16 =  2 MiB
  h16* qh  = (h16*)(ws + 16777216);                   // 8 MiB
  h16* kh  = (h16*)(ws + 25165824);                   // 8 MiB
  h16* vt  = (h16*)(ws + 33554432);                   // 8 MiB  [h][d][T]
  h16* yh  = (h16*)(ws + 41943040);                   // 8 MiB

  cvt_f32_f16<<<(4194304 + 255) / 256, 256, 0, stream>>>(x, xh, 4194304);
  cvt_f32_f16<<<(3145728 + 255) / 256, 256, 0, stream>>>(qkv_w, wh, 3145728);
  cvt_f32_f16<<<(1048576 + 255) / 256, 256, 0, stream>>>(proj_w, pwh, 1048576);

  qkv_kernel <<<dim3(64, 24), 128, 0, stream>>>(xh, wh, ve, lambdas, qh, kh, vt);
  attn_kernel<<<dim3(64, 8),  128, 0, stream>>>(qh, kh, vt, yh);
  proj_kernel<<<dim3(64, 8),  128, 0, stream>>>(yh, pwh, (float*)d_out);
}
